// SVDMamba_37641093382208
// MI455X (gfx1250) — matrix operand order repaired and checked by reading, not compiled, not run on hardware
//
#include <hip/hip_runtime.h>
#include <hip/hip_bf16.h>
#include <stdint.h>

// Flip to 0 if the gfx1250 assembler rejects the async mnemonics.
#define USE_ASYNC_LDS_COPY 1

typedef __bf16 bf16_t;
typedef __attribute__((ext_vector_type(16))) __bf16 v16bf;
typedef __attribute__((ext_vector_type(8)))  __bf16 v8bf;
typedef __attribute__((ext_vector_type(4)))  __bf16 v4bf;
typedef __attribute__((ext_vector_type(8)))  float  v8f;

#define D_MODEL 2048
#define D_INNER 4096
#define NTOK    8192   // B*S = 2*4096
#define LN_EPS  1e-5f

#define BM 128
#define BN 64
#define BK 32
#define LDAP 40   // padded LDS row stride in bf16 (32 data + 8 pad) -> conflict-free
#define NSTAGE 3  // ring depth: allows a single barrier per K-step

// ---------------- async global->LDS copy (CDNA5, ASYNCcnt-tracked) -------------
__device__ __forceinline__ void async_copy_b128(bf16_t* lds_dst, const bf16_t* g_src) {
#if USE_ASYNC_LDS_COPY
  // Generic shared-memory addr: low 32 bits are the LDS byte offset.
  unsigned l = (unsigned)(uintptr_t)lds_dst;
  asm volatile("global_load_async_to_lds_b128 %0, %1, off"
               :: "v"(l), "v"(g_src) : "memory");
#else
  *(v8bf*)lds_dst = *(const v8bf*)g_src;
#endif
}
// Wait until at most N async ops remain outstanding (async loads complete
// in order -> <=N outstanding means everything older than the newest N landed).
template <int N>
__device__ __forceinline__ void async_wait() {
#if USE_ASYNC_LDS_COPY
  asm volatile("s_wait_asynccnt %0" :: "n"(N) : "memory");
#endif
}

// ---------------- WMMA helpers -------------------------------------------------
// A fragment (16x32 bf16): lane m = row, kh = lane>>4 selects K-interleave:
//   elems 0..7  -> K = kh*8 + 0..7 ; elems 8..15 -> K = 16 + kh*8 + 0..7
__device__ __forceinline__ v16bf load_a_frag(const bf16_t* row_base, int kh) {
  const v8bf lo = *(const v8bf*)(row_base + kh * 8);
  const v8bf hi = *(const v8bf*)(row_base + 16 + kh * 8);
  v16bf f;
#pragma unroll
  for (int j = 0; j < 8; ++j) { f[j] = lo[j]; f[j + 8] = hi[j]; }
  return f;
}
// B fragment (32x16 bf16, B stored row-major NxK): lane n = row,
//   elems 0..7  -> K = kh*8 + 0..7 ; elems 8..15 -> K = 16 + kh*8 + 0..7
__device__ __forceinline__ v16bf load_b_frag(const bf16_t* row_base, int kh) {
  const v8bf lo = *(const v8bf*)(row_base + kh * 8);
  const v8bf hi = *(const v8bf*)(row_base + 16 + kh * 8);
  v16bf f;
#pragma unroll
  for (int j = 0; j < 8; ++j) { f[j] = lo[j]; f[j + 8] = hi[j]; }
  return f;
}
__device__ __forceinline__ v8f wmma_bf16(v16bf a, v16bf b, v8f c) {
  return __builtin_amdgcn_wmma_f32_16x16x32_bf16(false, a, false, b, (short)0, c,
                                                 false, false);
}
// sigmoid on a pre-clamped argument; v_rcp_f32 instead of IEEE division.
__device__ __forceinline__ float fast_sigmoid(float xc) {
  return __builtin_amdgcn_rcpf(1.0f + __expf(-xc));
}

// ---------------- LayerNorm -> bf16 -------------------------------------------
__device__ __forceinline__ float block_reduce_sum_256(float v, float* red) {
  const int t = threadIdx.x;
  red[t] = v;
  __syncthreads();
#pragma unroll
  for (int off = 128; off >= 1; off >>= 1) {
    if (t < off) red[t] += red[t + off];
    __syncthreads();
  }
  float r = red[0];
  __syncthreads();
  return r;
}

__global__ __launch_bounds__(256) void ln_to_bf16_kernel(
    const float* __restrict__ x, const float* __restrict__ gamma,
    const float* __restrict__ beta, bf16_t* __restrict__ xn) {
  __shared__ float red[256];
  const int row = blockIdx.x;
  const int t = threadIdx.x;
  const float4* xr = (const float4*)(x + (size_t)row * D_MODEL);
  float4 v[2];
  v[0] = xr[t];
  v[1] = xr[t + 256];
  float s = v[0].x + v[0].y + v[0].z + v[0].w + v[1].x + v[1].y + v[1].z + v[1].w;
  const float mu = block_reduce_sum_256(s, red) * (1.0f / D_MODEL);
  float q = 0.f;
#pragma unroll
  for (int i = 0; i < 2; ++i) {
    float dx;
    dx = v[i].x - mu; q += dx * dx;
    dx = v[i].y - mu; q += dx * dx;
    dx = v[i].z - mu; q += dx * dx;
    dx = v[i].w - mu; q += dx * dx;
  }
  const float var = block_reduce_sum_256(q, red) * (1.0f / D_MODEL);
  const float rs = rsqrtf(var + LN_EPS);
  bf16_t* xo = xn + (size_t)row * D_MODEL;
#pragma unroll
  for (int i = 0; i < 2; ++i) {
    const int c4 = t + i * 256;
    const float4 g = ((const float4*)gamma)[c4];
    const float4 b = ((const float4*)beta)[c4];
    v4bf o;
    o[0] = (bf16_t)((v[i].x - mu) * rs * g.x + b.x);
    o[1] = (bf16_t)((v[i].y - mu) * rs * g.y + b.y);
    o[2] = (bf16_t)((v[i].z - mu) * rs * g.z + b.z);
    o[3] = (bf16_t)((v[i].w - mu) * rs * g.w + b.w);
    ((v4bf*)xo)[c4] = o;
  }
}

// ---------------- fp32 -> bf16 weight conversion -------------------------------
__global__ __launch_bounds__(256) void f32_to_bf16_kernel(
    const float* __restrict__ src, bf16_t* __restrict__ dst, int n4) {
  const int i = blockIdx.x * blockDim.x + threadIdx.x;
  if (i < n4) {
    const float4 v = ((const float4*)src)[i];
    v4bf o;
    o[0] = (bf16_t)v.x; o[1] = (bf16_t)v.y; o[2] = (bf16_t)v.z; o[3] = (bf16_t)v.w;
    ((v4bf*)dst)[i] = o;
  }
}

// ---------------- softmax(-dt) over D_INNER ------------------------------------
__global__ __launch_bounds__(512) void dt_softmax_kernel(
    const float* __restrict__ dt, float* __restrict__ w) {
  __shared__ float red[512];
  const int t = threadIdx.x;
  float v[8];
  float mx = -3.4e38f;
#pragma unroll
  for (int i = 0; i < 8; ++i) { v[i] = -dt[t + i * 512]; mx = fmaxf(mx, v[i]); }
  red[t] = mx;
  __syncthreads();
  for (int off = 256; off >= 1; off >>= 1) {
    if (t < off) red[t] = fmaxf(red[t], red[t + off]);
    __syncthreads();
  }
  const float m = red[0];
  __syncthreads();
  float s = 0.f;
#pragma unroll
  for (int i = 0; i < 8; ++i) { v[i] = __expf(v[i] - m); s += v[i]; }
  red[t] = s;
  __syncthreads();
  for (int off = 256; off >= 1; off >>= 1) {
    if (t < off) red[t] += red[t + off];
    __syncthreads();
  }
  const float inv = 1.0f / red[0];
#pragma unroll
  for (int i = 0; i < 8; ++i) w[t + i * 512] = v[i] * inv;
}

// ---------------- GEMM1: act = f(xn @ W_in^T) fused gating ---------------------
// A [NTOK, D_MODEL] bf16, Wz/Wg [D_INNER, D_MODEL] bf16, act [NTOK, D_INNER] bf16
// 3-stage LDS ring fed by async copies; one barrier per K-step; the pipeline is
// structured so every ring index is a compile-time constant (no mid-body exits,
// so accumulators stay pinned in their WMMA registers).
__global__ __launch_bounds__(256) void gemm1_fused_kernel(
    const bf16_t* __restrict__ A, const bf16_t* __restrict__ Wz,
    const bf16_t* __restrict__ Wg, const float* __restrict__ dtw,
    bf16_t* __restrict__ act) {
  __shared__ __align__(16) bf16_t sA[NSTAGE][BM * LDAP];
  __shared__ __align__(16) bf16_t sBz[NSTAGE][BN * LDAP];
  __shared__ __align__(16) bf16_t sBg[NSTAGE][BN * LDAP];

  const int t = threadIdx.x;
  const int lane = t & 31;
  const int wv = t >> 5;
  const int wm = wv & 3;   // wave M index 0..3  (32 rows each)
  const int wn = wv >> 2;  // wave N index 0..1  (32 cols each)
  const int m0 = blockIdx.x * BM;
  const int n0 = blockIdx.y * BN;

  v8f accz[2][2] = {};
  v8f accg[2][2] = {};

  const int ar = t >> 2;        // loader row (A adds +64 second chunk; B: 0..63)
  const int ac = (t & 3) * 8;   // loader col chunk (8 bf16 = 16B)
  const int rsel = lane & 15;
  const int kh = lane >> 4;

  // 4 async instructions per tile batch (per wave)
  auto issue_tile = [&](int buf, int k0) {
    async_copy_b128(&sA[buf][ar * LDAP + ac],
                    A + (size_t)(m0 + ar) * D_MODEL + k0 + ac);
    async_copy_b128(&sA[buf][(ar + 64) * LDAP + ac],
                    A + (size_t)(m0 + ar + 64) * D_MODEL + k0 + ac);
    async_copy_b128(&sBz[buf][ar * LDAP + ac],
                    Wz + (size_t)(n0 + ar) * D_MODEL + k0 + ac);
    async_copy_b128(&sBg[buf][ar * LDAP + ac],
                    Wg + (size_t)(n0 + ar) * D_MODEL + k0 + ac);
  };
  auto compute_tile = [&](int buf) {
    v16bf af[2], bzf[2], bgf[2];
    af[0]  = load_a_frag(&sA[buf][(wm * 32 + rsel) * LDAP], kh);
    af[1]  = load_a_frag(&sA[buf][(wm * 32 + 16 + rsel) * LDAP], kh);
    bzf[0] = load_b_frag(&sBz[buf][(wn * 32 + rsel) * LDAP], kh);
    bzf[1] = load_b_frag(&sBz[buf][(wn * 32 + 16 + rsel) * LDAP], kh);
    bgf[0] = load_b_frag(&sBg[buf][(wn * 32 + rsel) * LDAP], kh);
    bgf[1] = load_b_frag(&sBg[buf][(wn * 32 + 16 + rsel) * LDAP], kh);
#pragma unroll
    for (int tm = 0; tm < 2; ++tm)
#pragma unroll
      for (int tn = 0; tn < 2; ++tn) {
        accz[tm][tn] = wmma_bf16(af[tm], bzf[tn], accz[tm][tn]);
        accg[tm][tn] = wmma_bf16(af[tm], bgf[tn], accg[tm][tn]);
      }
  };
  auto step = [&](int k, int cur, int nxt) {
    issue_tile(nxt, (k + 1) * BK);
    async_wait<4>();   // tile k landed (only tile k+1's ops may be pending)
    __syncthreads();   // tile-k data visible everywhere; WAR-protects the ring
    compute_tile(cur);
  };

  // T = 64 tiles; 63 steady steps = 21 groups x 3 constant ring phases.
  issue_tile(0, 0);
#pragma unroll 1
  for (int g = 0; g < 21; ++g) {
    const int k = g * 3;
    step(k + 0, 0, 1);
    step(k + 1, 1, 2);
    step(k + 2, 2, 0);
  }
  async_wait<0>();
  __syncthreads();
  compute_tile(0);  // tile 63 lives in buffer 63%3 == 0

  // Epilogue: z*sigmoid(clip z) * softmax(-dt)[n] * silu(clip gate)
  const int rhi = (lane >> 4) * 8;
#pragma unroll
  for (int tm = 0; tm < 2; ++tm)
#pragma unroll
    for (int tn = 0; tn < 2; ++tn) {
      const int n = n0 + wn * 32 + tn * 16 + rsel;
      const float dw = dtw[n];
#pragma unroll
      for (int i = 0; i < 8; ++i) {
        const int m = m0 + wm * 32 + tm * 16 + rhi + i;
        const float z = accz[tm][tn][i];
        const float g = accg[tm][tn][i];
        const float zc = fminf(fmaxf(z, -15.f), 15.f);
        const float gc = fminf(fmaxf(g, -15.f), 15.f);
        const float val = z * fast_sigmoid(zc) * dw * (gc * fast_sigmoid(gc));
        act[(size_t)m * D_INNER + n] = (bf16_t)val;
      }
    }
}

// ---------------- GEMM2: out = act @ W_out^T + x -------------------------------
// A [NTOK, D_INNER] bf16, Bw [D_MODEL, D_INNER] bf16, out [NTOK, D_MODEL] fp32
__global__ __launch_bounds__(256) void gemm2_res_kernel(
    const bf16_t* __restrict__ A, const bf16_t* __restrict__ Bw,
    const float* __restrict__ resid, float* __restrict__ out) {
  __shared__ __align__(16) bf16_t sA[NSTAGE][BM * LDAP];
  __shared__ __align__(16) bf16_t sB[NSTAGE][BN * LDAP];

  const int t = threadIdx.x;
  const int lane = t & 31;
  const int wv = t >> 5;
  const int wm = wv & 3;
  const int wn = wv >> 2;
  const int m0 = blockIdx.x * BM;
  const int n0 = blockIdx.y * BN;

  v8f acc[2][2] = {};

  const int ar = t >> 2;
  const int ac = (t & 3) * 8;
  const int rsel = lane & 15;
  const int kh = lane >> 4;

  // 3 async instructions per tile batch (per wave)
  auto issue_tile = [&](int buf, int k0) {
    async_copy_b128(&sA[buf][ar * LDAP + ac],
                    A + (size_t)(m0 + ar) * D_INNER + k0 + ac);
    async_copy_b128(&sA[buf][(ar + 64) * LDAP + ac],
                    A + (size_t)(m0 + ar + 64) * D_INNER + k0 + ac);
    async_copy_b128(&sB[buf][ar * LDAP + ac],
                    Bw + (size_t)(n0 + ar) * D_INNER + k0 + ac);
  };
  auto compute_tile = [&](int buf) {
    v16bf af[2], bfr[2];
    af[0]  = load_a_frag(&sA[buf][(wm * 32 + rsel) * LDAP], kh);
    af[1]  = load_a_frag(&sA[buf][(wm * 32 + 16 + rsel) * LDAP], kh);
    bfr[0] = load_b_frag(&sB[buf][(wn * 32 + rsel) * LDAP], kh);
    bfr[1] = load_b_frag(&sB[buf][(wn * 32 + 16 + rsel) * LDAP], kh);
#pragma unroll
    for (int tm = 0; tm < 2; ++tm)
#pragma unroll
      for (int tn = 0; tn < 2; ++tn)
        acc[tm][tn] = wmma_bf16(af[tm], bfr[tn], acc[tm][tn]);
  };
  auto step = [&](int k, int cur, int nxt) {
    issue_tile(nxt, (k + 1) * BK);
    async_wait<3>();
    __syncthreads();
    compute_tile(cur);
  };

  // T = 128 tiles; 127 steady steps = 1 peeled + 42 groups x 3 constant phases.
  issue_tile(0, 0);
  step(0, 0, 1);
#pragma unroll 1
  for (int g = 0; g < 42; ++g) {
    const int k = 1 + g * 3;
    step(k + 0, 1, 2);
    step(k + 1, 2, 0);
    step(k + 2, 0, 1);
  }
  async_wait<0>();
  __syncthreads();
  compute_tile(1);  // tile 127 lives in buffer 127%3 == 1

  const int rhi = (lane >> 4) * 8;
#pragma unroll
  for (int tm = 0; tm < 2; ++tm)
#pragma unroll
    for (int tn = 0; tn < 2; ++tn) {
      const int n = n0 + wn * 32 + tn * 16 + rsel;
#pragma unroll
      for (int i = 0; i < 8; ++i) {
        const int m = m0 + wm * 32 + tm * 16 + rhi + i;
        out[(size_t)m * D_MODEL + n] = acc[tm][tn][i] + resid[(size_t)m * D_MODEL + n];
      }
    }
}

// ---------------- host launcher ------------------------------------------------
extern "C" void kernel_launch(void* const* d_in, const int* in_sizes, int n_in,
                              void* d_out, int out_size, void* d_ws, size_t ws_size,
                              hipStream_t stream) {
  const float* x     = (const float*)d_in[0];  // [2,4096,2048]
  const float* gamma = (const float*)d_in[1];  // [2048]
  const float* beta  = (const float*)d_in[2];  // [2048]
  const float* W_in  = (const float*)d_in[3];  // [8192,2048]
  const float* W_out = (const float*)d_in[4];  // [2048,4096]
  const float* dt    = (const float*)d_in[5];  // [4096]
  float* out = (float*)d_out;

  char* ws = (char*)d_ws;
  bf16_t* xn_bf   = (bf16_t*)(ws + 0);           // 8192*2048*2  = 32 MiB
  bf16_t* win_bf  = (bf16_t*)(ws + 33554432);    // 8192*2048*2  = 32 MiB
  bf16_t* wout_bf = (bf16_t*)(ws + 67108864);    // 2048*4096*2  = 16 MiB
  bf16_t* act_bf  = (bf16_t*)(ws + 83886080);    // 8192*4096*2  = 64 MiB
  float*  dtw     = (float*) (ws + 150994944);   // 4096*4

  ln_to_bf16_kernel<<<NTOK, 256, 0, stream>>>(x, gamma, beta, xn_bf);

  const int n4_win  = 2 * D_INNER * D_MODEL / 4;
  const int n4_wout = D_MODEL * D_INNER / 4;
  f32_to_bf16_kernel<<<(n4_win + 255) / 256, 256, 0, stream>>>(W_in, win_bf, n4_win);
  f32_to_bf16_kernel<<<(n4_wout + 255) / 256, 256, 0, stream>>>(W_out, wout_bf, n4_wout);

  dt_softmax_kernel<<<1, 512, 0, stream>>>(dt, dtw);

  gemm1_fused_kernel<<<dim3(NTOK / BM, D_INNER / BN), 256, 0, stream>>>(
      xn_bf, win_bf, win_bf + (size_t)D_INNER * D_MODEL, dtw, act_bf);

  gemm2_res_kernel<<<dim3(NTOK / BM, D_MODEL / BN), 256, 0, stream>>>(
      act_bf, wout_bf, x, out);
}
